// MLA_82454782148667
// MI455X (gfx1250) — matrix operand order repaired and checked by reading, not compiled, not run on hardware
//
#include <hip/hip_runtime.h>

typedef __attribute__((ext_vector_type(16))) __bf16 v16bf;
typedef __attribute__((ext_vector_type(8)))  __bf16 v8bf;
typedef __attribute__((ext_vector_type(8)))  float  v8f;

#define NH     16
#define DQK    192
#define DNOPE  128
#define DROPE  64
#define DV     128
#define RANK   512
#define DIMX   2048
#define SEQ    2048
#define BATCH  2
#define NTOK   (BATCH * SEQ)

// ---------------- bf16 helpers (RNE convert) ----------------
__device__ __forceinline__ __bf16 to_bf16(float f) {
    unsigned u = __float_as_uint(f);
    u += 0x7FFFu + ((u >> 16) & 1u);
    unsigned short s = (unsigned short)(u >> 16);
    __bf16 r;
    __builtin_memcpy(&r, &s, 2);
    return r;
}
__device__ __forceinline__ float bf16_to_f32(__bf16 b) {
    unsigned short s;
    __builtin_memcpy(&s, &b, 2);
    return __uint_as_float((unsigned)s << 16);
}

// ---------------- WMMA fragment loaders (per CDNA5 ISA layouts) ----------------
union ABf { v16bf v; v8bf h[2]; };

// A matrix 16x32 bf16, row-major source: lane holds row (lane&15),
// K = kbase+{0..7} and kbase+16+{0..7} where kbase = (lane>>4)*8.
__device__ __forceinline__ v16bf load_a_frag(const __bf16* A, int lda, int k0, int lane) {
    int r = lane & 15, kb = (lane >> 4) * 8;
    const __bf16* p = A + (long)r * lda + k0 + kb;
    ABf f;
    f.h[0] = *(const v8bf*)p;
    f.h[1] = *(const v8bf*)(p + 16);
    return f.v;
}

// B matrix 32x16 bf16 from a [N,K]-row-major source (i.e. B^T storage):
// lane holds column N = n0+(lane&15), K = kbase+{0..7} and kbase+16+{0..7} where kbase=(lane>>4)*8.
__device__ __forceinline__ v16bf load_b_frag(const __bf16* BT, int ldb, int n0, int k0, int lane) {
    int n = lane & 15, kb = (lane >> 4) * 8;
    const __bf16* p = BT + (long)(n0 + n) * ldb + k0 + kb;
    ABf f;
    f.h[0] = *(const v8bf*)p;
    f.h[1] = *(const v8bf*)(p + 16);
    return f.v;
}

__device__ __forceinline__ v8f wmma_bf16(v16bf a, v16bf b, v8f c) {
    return __builtin_amdgcn_wmma_f32_16x16x32_bf16(false, a, false, b, (short)0, c, false, false);
}

// ---------------- elementwise / transform kernels ----------------
__global__ void cvt_f32_bf16_kernel(const float* __restrict__ in, __bf16* __restrict__ out, long n) {
    long i = (long)blockIdx.x * blockDim.x + threadIdx.x;
    if (i < n) out[i] = to_bf16(in[i]);
}

// W[K,N] f32 -> WT[N,K] bf16, LDS-tiled (K and N multiples of 32)
__global__ __launch_bounds__(256) void wtrans_kernel(const float* __restrict__ W,
                                                     __bf16* __restrict__ WT, int K, int N) {
    __shared__ __bf16 tile[32][33];
    int kt = blockIdx.y << 5, nt = blockIdx.x << 5;
    int tx = threadIdx.x & 31, ty = threadIdx.x >> 5;
    #pragma unroll
    for (int r = 0; r < 32; r += 8)
        tile[ty + r][tx] = to_bf16(W[(long)(kt + ty + r) * N + nt + tx]);
    __syncthreads();
    #pragma unroll
    for (int r = 0; r < 32; r += 8)
        WT[(long)(nt + ty + r) * K + kt + tx] = tile[tx][ty + r];
}

// rope on q[:, h, 128:192] and kc[:, 512:576], in-place bf16
__global__ void rope_kernel(__bf16* __restrict__ q, __bf16* __restrict__ kc,
                            const int* __restrict__ start_pos) {
    const long QTOT = (long)NTOK * NH * (DROPE / 2);
    const long KTOT = (long)NTOK * (DROPE / 2);
    long idx = (long)blockIdx.x * blockDim.x + threadIdx.x;
    if (idx >= QTOT + KTOT) return;
    int i, t;
    __bf16* p;
    if (idx < QTOT) {
        long tmp = idx;
        i = (int)(tmp & 31); tmp >>= 5;
        int h = (int)(tmp & 15); t = (int)(tmp >> 4);
        p = q + (long)t * (NH * DQK) + h * DQK + DNOPE + 2 * i;
    } else {
        long tmp = idx - QTOT;
        i = (int)(tmp & 31); t = (int)(tmp >> 5);
        p = kc + (long)t * (RANK + DROPE) + RANK + 2 * i;
    }
    int pos = start_pos[0] + (t & (SEQ - 1));
    float theta = __expf(-((float)(2 * i) / 64.0f) * 9.210340371976184f); // ROPE_BASE^(-2i/64)
    float ang = (float)pos * theta;
    float c = __cosf(ang), s = __sinf(ang);
    float x0 = bf16_to_f32(p[0]), x1 = bf16_to_f32(p[1]);
    p[0] = to_bf16(x0 * c - x1 * s);
    p[1] = to_bf16(x1 * c + x0 * s);
}

// k_full[tok, h, 0:128] = k_nope ; k_full[tok, h, 128:192] = roped kc rope part
__global__ void kfull_kernel(const __bf16* __restrict__ k_nope, const __bf16* __restrict__ kc,
                             __bf16* __restrict__ kf) {
    long idx = (long)blockIdx.x * blockDim.x + threadIdx.x;
    if (idx >= (long)NTOK * NH * DQK) return;
    int d = (int)(idx % DQK);
    long th = idx / DQK;
    int h = (int)(th % NH);
    long t = th / NH;
    kf[idx] = (d < DNOPE) ? k_nope[t * (NH * DNOPE) + h * DNOPE + d]
                          : kc[t * (RANK + DROPE) + RANK + (d - DNOPE)];
}

// v [b*2048+t, c=h*128+d] -> vT [b*2048+c, t]  (per-batch 2048x2048 transpose, LDS tiled)
__global__ __launch_bounds__(256) void vtrans_kernel(const __bf16* __restrict__ v,
                                                     __bf16* __restrict__ vT) {
    __shared__ __bf16 tile[32][33];
    int b = blockIdx.z;
    int t0 = blockIdx.y << 5, c0 = blockIdx.x << 5;
    int tx = threadIdx.x & 31, ty = threadIdx.x >> 5;
    #pragma unroll
    for (int r = 0; r < 32; r += 8)
        tile[ty + r][tx] = v[((long)b * SEQ + t0 + ty + r) * (NH * DV) + c0 + tx];
    __syncthreads();
    #pragma unroll
    for (int r = 0; r < 32; r += 8)
        vT[((long)b * (NH * DV) + c0 + ty + r) * SEQ + t0 + tx] = tile[tx][ty + r];
}

// ---------------- generic bf16 WMMA GEMM ----------------
// Wave tile 32x64, ping-pong double buffering (K % 64 == 0 for all uses here),
// so no register rotation copies: each WMMA consumes the registers its loads
// landed in, and waits become partial (older load group only).
// C[M,N] = A[M,K](lda) * BT[N,K]^T ; bf16 out (Cb) or f32 out + bias (Cf)
__global__ __launch_bounds__(128) void gemm_bf16_kernel(
    const __bf16* __restrict__ A, int lda,
    const __bf16* __restrict__ BT,
    __bf16* __restrict__ Cb, float* __restrict__ Cf,
    const float* __restrict__ bias, int ldc,
    int M, int N, int K) {
    int lane = threadIdx.x & 31, widx = threadIdx.x >> 5;
    int tilesN = N >> 6;
    int id = blockIdx.x * 4 + widx;
    if (id >= (M >> 5) * tilesN) return;
    int m0 = (id / tilesN) << 5;
    int n0 = (id % tilesN) << 6;
    const __bf16* A0 = A + (long)m0 * lda;
    const __bf16* A1 = A0 + (long)16 * lda;

    v8f acc[2][4] = {{{0},{0},{0},{0}},{{0},{0},{0},{0}}};

    v16bf a0[2], a1[2], b[2][4];
    // prologue: buffer 0 <- k = 0
    a0[0] = load_a_frag(A0, lda, 0, lane);
    a1[0] = load_a_frag(A1, lda, 0, lane);
    #pragma unroll
    for (int t = 0; t < 4; t++) b[0][t] = load_b_frag(BT, K, n0 + t * 16, 0, lane);

    for (int k = 0; k < K; k += 64) {
        // buffer 1 <- k + 32 (always in range: K % 64 == 0)
        a0[1] = load_a_frag(A0, lda, k + 32, lane);
        a1[1] = load_a_frag(A1, lda, k + 32, lane);
        #pragma unroll
        for (int t = 0; t < 4; t++) b[1][t] = load_b_frag(BT, K, n0 + t * 16, k + 32, lane);
        if (k + 64 < K)
            __builtin_prefetch(A0 + (long)(lane & 15) * lda + k + 64, 0, 1);
        // compute buffer 0 (k) while buffer 1 loads are in flight
        #pragma unroll
        for (int t = 0; t < 4; t++) acc[0][t] = wmma_bf16(a0[0], b[0][t], acc[0][t]);
        #pragma unroll
        for (int t = 0; t < 4; t++) acc[1][t] = wmma_bf16(a1[0], b[0][t], acc[1][t]);
        // buffer 0 <- k + 64
        if (k + 64 < K) {
            a0[0] = load_a_frag(A0, lda, k + 64, lane);
            a1[0] = load_a_frag(A1, lda, k + 64, lane);
            #pragma unroll
            for (int t = 0; t < 4; t++) b[0][t] = load_b_frag(BT, K, n0 + t * 16, k + 64, lane);
        }
        // compute buffer 1 (k + 32) while buffer 0 loads are in flight
        #pragma unroll
        for (int t = 0; t < 4; t++) acc[0][t] = wmma_bf16(a0[1], b[1][t], acc[0][t]);
        #pragma unroll
        for (int t = 0; t < 4; t++) acc[1][t] = wmma_bf16(a1[1], b[1][t], acc[1][t]);
    }
    int hi = lane >> 4, col = lane & 15;
    #pragma unroll
    for (int mi = 0; mi < 2; mi++)
        #pragma unroll
        for (int t = 0; t < 4; t++)
            #pragma unroll
            for (int r = 0; r < 8; r++) {
                long idx = (long)(m0 + mi * 16 + r + 8 * hi) * ldc + n0 + t * 16 + col;
                if (Cb) Cb[idx] = to_bf16(acc[mi][t][r]);
                else    Cf[idx] = acc[mi][t][r] + (bias ? bias[n0 + t * 16 + col] : 0.0f);
            }
}

// ---------------- flash attention (causal, online softmax, all-WMMA) ----------------
// q/kf: [tok, H, 192] bf16 ; vT: [b,h,dv,s] bf16 ; cv: [tok, H*128] bf16
__global__ __launch_bounds__(128) void attn_kernel(
    const __bf16* __restrict__ q, const __bf16* __restrict__ kf,
    const __bf16* __restrict__ vT, __bf16* __restrict__ cv) {
    __shared__ __align__(16) __bf16 plds[4][16][32];
    int lane = threadIdx.x & 31, widx = threadIdx.x >> 5;
    int g = blockIdx.x * 4 + widx;             // 4096 wave-tiles total
    int qt = g & 127;                          // 128 q-tiles per (b,h)
    int bh = g >> 7;
    int h = bh & 15, b = bh >> 4;
    int q0 = qt << 4;
    int hi = lane >> 4, col = lane & 15;

    const __bf16* qbase = q  + ((long)(b * SEQ + q0)) * (NH * DQK) + h * DQK;
    const __bf16* kbase = kf + ((long)(b * SEQ)) * (NH * DQK) + h * DQK;
    const __bf16* vbase = vT + ((long)(b * NH + h) * DV) * SEQ;

    // preload q fragments (reused across every key chunk)
    v16bf qa[6];
    #pragma unroll
    for (int i = 0; i < 6; i++) qa[i] = load_a_frag(qbase, NH * DQK, i * 32, lane);

    v8f o[8] = {{0},{0},{0},{0},{0},{0},{0},{0}};
    float m[8], l[8];
    #pragma unroll
    for (int r = 0; r < 8; r++) { m[r] = -1e30f; l[r] = 0.0f; }

    const float scale = 0.0721687836487032f;   // 1/sqrt(192)
    int kend = q0 + 16;
    for (int kc0 = 0; kc0 < kend; kc0 += 32) {
        // Issue K fragments first, then V fragments: LOADcnt completes in order,
        // so the S WMMAs only wait for the K loads and the V loads drain behind
        // the softmax + LDS round-trip.
        v16bf kb[12];
        #pragma unroll
        for (int i = 0; i < 6; i++) {
            kb[2 * i]     = load_b_frag(kbase, NH * DQK, kc0,      i * 32, lane);
            kb[2 * i + 1] = load_b_frag(kbase, NH * DQK, kc0 + 16, i * 32, lane);
        }
        v16bf bv[8];
        #pragma unroll
        for (int t = 0; t < 8; t++) bv[t] = load_b_frag(vbase, SEQ, t * 16, kc0, lane);
        if (kc0 + 32 < kend)  // prefetch next chunk's K rows (one row per lane)
            __builtin_prefetch(kbase + (long)(kc0 + 32 + lane) * (NH * DQK), 0, 1);

        // S = q @ K^T for a 16x32 score tile
        v8f s0 = {0}, s1 = {0};
        #pragma unroll
        for (int i = 0; i < 6; i++) {
            s0 = wmma_bf16(qa[i], kb[2 * i],     s0);
            s1 = wmma_bf16(qa[i], kb[2 * i + 1], s1);
        }
        // online softmax (C-layout: lane=col, element r = row r+8*hi)
        float alpha[8];
        #pragma unroll
        for (int r = 0; r < 8; r++) {
            int qg = q0 + r + 8 * hi;
            float e0 = s0[r] * scale; if (kc0 + col      > qg) e0 = -1e30f;
            float e1 = s1[r] * scale; if (kc0 + 16 + col > qg) e1 = -1e30f;
            float mx = fmaxf(e0, e1);
            #pragma unroll
            for (int off = 8; off > 0; off >>= 1) mx = fmaxf(mx, __shfl_xor(mx, off, 32));
            float mn = fmaxf(m[r], mx);
            float a  = __expf(m[r] - mn);
            float p0 = __expf(e0 - mn);
            float p1 = __expf(e1 - mn);
            float sum = p0 + p1;
            #pragma unroll
            for (int off = 8; off > 0; off >>= 1) sum += __shfl_xor(sum, off, 32);
            l[r] = l[r] * a + sum;
            m[r] = mn;
            alpha[r] = a;
            // stage P (C-layout -> LDS) for the A-fragment reload
            plds[widx][r + 8 * hi][col]      = to_bf16(p0);
            plds[widx][r + 8 * hi][16 + col] = to_bf16(p1);
        }
        #pragma unroll
        for (int t = 0; t < 8; t++)       // rescale overlaps the LDS round-trip
            #pragma unroll
            for (int r = 0; r < 8; r++) o[t][r] *= alpha[r];
        // P (16x32) as A-fragment from LDS, then o += P @ V
        v16bf pa = load_a_frag(&plds[widx][0][0], 32, 0, lane);
        #pragma unroll
        for (int t = 0; t < 8; t++) o[t] = wmma_bf16(pa, bv[t], o[t]);
    }
    // epilogue: o /= l, store bf16
    #pragma unroll
    for (int t = 0; t < 8; t++)
        #pragma unroll
        for (int r = 0; r < 8; r++) {
            int row = r + 8 * hi;
            float val = o[t][r] / l[r];
            cv[(long)(b * SEQ + q0 + row) * (NH * DV) + h * DV + t * 16 + col] = to_bf16(val);
        }
}

// ---------------- host launcher ----------------
static inline int ceil_div(long a, long b) { return (int)((a + b - 1) / b); }

extern "C" void kernel_launch(void* const* d_in, const int* in_sizes, int n_in,
                              void* d_out, int out_size, void* d_ws, size_t ws_size,
                              hipStream_t stream) {
    const float* x        = (const float*)d_in[0];
    const int*   startpos = (const int*)d_in[1];
    const float* wq       = (const float*)d_in[2];
    const float* wkc      = (const float*)d_in[3];
    const float* wvc      = (const float*)d_in[4];
    const float* wku      = (const float*)d_in[5];
    const float* wvu      = (const float*)d_in[6];
    const float* wo       = (const float*)d_in[7];
    const float* wob      = (const float*)d_in[8];
    float* out = (float*)d_out;

    char* cur = (char*)d_ws;
    auto alloc = [&](size_t elems) {
        __bf16* r = (__bf16*)cur;
        cur += ((elems * 2) + 255) & ~(size_t)255;
        return r;
    };
    __bf16* xb    = alloc((size_t)NTOK * DIMX);
    __bf16* wqT   = alloc((size_t)(NH * DQK) * DIMX);
    __bf16* wkcT  = alloc((size_t)(RANK + DROPE) * DIMX);
    __bf16* wvcT  = alloc((size_t)RANK * DIMX);
    __bf16* wkuT  = alloc((size_t)(NH * DNOPE) * RANK);
    __bf16* wvuT  = alloc((size_t)(NH * DV) * RANK);
    __bf16* woT   = alloc((size_t)DIMX * (NH * DV));
    __bf16* qb    = alloc((size_t)NTOK * NH * DQK);
    __bf16* kcb   = alloc((size_t)NTOK * (RANK + DROPE));
    __bf16* vcb   = alloc((size_t)NTOK * RANK);
    __bf16* knope = alloc((size_t)NTOK * NH * DNOPE);
    __bf16* vb    = alloc((size_t)NTOK * NH * DV);
    __bf16* kful  = alloc((size_t)NTOK * NH * DQK);
    __bf16* vTb   = alloc((size_t)NTOK * NH * DV);
    __bf16* cvb   = alloc((size_t)NTOK * NH * DV);

    const int TB = 256;
    // 1. convert x
    cvt_f32_bf16_kernel<<<ceil_div((long)NTOK * DIMX, TB), TB, 0, stream>>>(x, xb, (long)NTOK * DIMX);
    // 2. transpose+convert weights -> [N,K] bf16 (LDS-tiled, all dims % 32 == 0)
    auto wtrans = [&](const float* W, __bf16* WT, int K, int N) {
        dim3 grid(N >> 5, K >> 5, 1);
        wtrans_kernel<<<grid, 256, 0, stream>>>(W, WT, K, N);
    };
    wtrans(wq,  wqT,  DIMX, NH * DQK);
    wtrans(wkc, wkcT, DIMX, RANK + DROPE);
    wtrans(wvc, wvcT, DIMX, RANK);
    wtrans(wku, wkuT, RANK, NH * DNOPE);
    wtrans(wvu, wvuT, RANK, NH * DV);
    wtrans(wo,  woT,  NH * DV, DIMX);

    auto gemm = [&](const __bf16* A, int lda, const __bf16* BT,
                    __bf16* Cb, float* Cf, const float* bias, int ldc,
                    int M, int N, int K) {
        int tiles = (M >> 5) * (N >> 6);
        gemm_bf16_kernel<<<ceil_div(tiles, 4), 128, 0, stream>>>(A, lda, BT, Cb, Cf, bias, ldc, M, N, K);
    };
    // 3-5. projections
    gemm(xb, DIMX, wqT,  qb,  nullptr, nullptr, NH * DQK,     NTOK, NH * DQK,     DIMX);
    gemm(xb, DIMX, wkcT, kcb, nullptr, nullptr, RANK + DROPE, NTOK, RANK + DROPE, DIMX);
    gemm(xb, DIMX, wvcT, vcb, nullptr, nullptr, RANK,         NTOK, RANK,         DIMX);
    // 6. rope
    {
        long n = (long)NTOK * NH * (DROPE / 2) + (long)NTOK * (DROPE / 2);
        rope_kernel<<<ceil_div(n, TB), TB, 0, stream>>>(qb, kcb, startpos);
    }
    // 7-8. up-projections (kc_nope is cols [0,512) of kcb, lda = 576)
    gemm(kcb, RANK + DROPE, wkuT, knope, nullptr, nullptr, NH * DNOPE, NTOK, NH * DNOPE, RANK);
    gemm(vcb, RANK,         wvuT, vb,    nullptr, nullptr, NH * DV,    NTOK, NH * DV,    RANK);
    // 9-10. assemble k_full and v^T
    kfull_kernel<<<ceil_div((long)NTOK * NH * DQK, TB), TB, 0, stream>>>(knope, kcb, kful);
    {
        dim3 grid((NH * DV) >> 5, SEQ >> 5, BATCH);
        vtrans_kernel<<<grid, 256, 0, stream>>>(vb, vTb);
    }
    // 11. flash attention: 4096 wave-tiles, 4 waves/block
    attn_kernel<<<1024, 128, 0, stream>>>(qb, kful, vTb, cvb);
    // 12. output projection with bias -> f32 d_out
    gemm(cvb, NH * DV, woT, nullptr, out, wob, DIMX, NTOK, DIMX, NH * DV);
    (void)in_sizes; (void)n_in; (void)out_size; (void)ws_size;
}
